// _AttentionModule_3015067042404
// MI455X (gfx1250) — matrix operand order repaired and checked by reading, not compiled, not run on hardware
//
#include <hip/hip_runtime.h>

typedef __attribute__((ext_vector_type(16))) _Float16 v16h;
typedef __attribute__((ext_vector_type(8)))  _Float16 v8h;
typedef __attribute__((ext_vector_type(8)))  float    v8f;

#define NB    8      // batch
#define CIN   256
#define ICH   128
#define HW    4096   // 64*64
#define MP    1024   // pooled 32*32
#define LDP   40     // padded LDS row stride (halves): 80B, 16B-aligned rows

static __device__ inline v16h cat16(v8h lo, v8h hi) {
  return __builtin_shufflevector(lo, hi, 0,1,2,3,4,5,6,7,8,9,10,11,12,13,14,15);
}

// A fragment (16x32 f16) from row-major storage: element (m,k) at p[m*ld + k]
// lanes 0-15 hold K 0..15, lanes 16-31 hold K 16..31, m = lane&15
static __device__ inline v16h afrag_rm(const _Float16* p, int ld, int lane) {
  int m = lane & 15;
  int d = (lane >> 4) << 4;
  const _Float16* r = p + (size_t)m * ld;
  v8h lo = *(const v8h*)(r + d);
  v8h hi = *(const v8h*)(r + 8 + d);
  return cat16(lo, hi);
}

// B fragment (32x16 f16) from N-major storage: element (k,n) at p[n*ld + k]
// lanes 0-15 hold K 0..15, lanes 16-31 hold K 16..31, n = lane&15
static __device__ inline v16h bfrag_nm(const _Float16* p, int ld, int lane) {
  int n = lane & 15;
  int d = (lane >> 4) << 4;
  return *(const v16h*)(p + (size_t)n * ld + d);
}

static __device__ inline v8f wmma16(v16h a, v16h b, v8f c) {
  return __builtin_amdgcn_wmma_f32_16x16x32_f16(false, a, false, b, (short)0, c,
                                                false, false);
}

// ---------------- weight fp32 -> f16 conversion ----------------
__global__ __launch_bounds__(256) void cvt_w_kernel(
    const float* __restrict__ gw, const float* __restrict__ tw,
    const float* __restrict__ pw, const float* __restrict__ Ww,
    _Float16* __restrict__ gh, _Float16* __restrict__ th,
    _Float16* __restrict__ ph, _Float16* __restrict__ Wh) {
  int t = blockIdx.x * 256 + threadIdx.x;
  if (t < ICH * CIN) {
    gh[t] = (_Float16)gw[t];
    th[t] = (_Float16)tw[t];
    ph[t] = (_Float16)pw[t];
    Wh[t] = (_Float16)Ww[t];   // 256*128 == 128*256 elements
  }
}

// ---------------- generic 1x1-conv projection: out(n,i) = X^T W^T + b -------
// Block: 128 spatial positions, 8 waves, each wave one 16n x 128i strip.
// Full f16 weight matrix (64 KB) staged in LDS once; activation tile staged
// per K-chunk n-major with padded stride -> inner loop is pure LDS + WMMA.
__global__ __launch_bounds__(256) void proj_kernel(
    const float* __restrict__ x, const _Float16* __restrict__ wh,
    const float* __restrict__ bias, _Float16* __restrict__ outm) {
  __shared__ v16h lw_v[ICH * CIN / 16];   // 64 KB weights (i, c), 32B aligned
  __shared__ v8h  lx_v[128 * LDP / 8];    // 10 KB activation tile (n, c)
  _Float16* lw = (_Float16*)lw_v;
  _Float16* lx = (_Float16*)lx_v;
  const int n0   = blockIdx.x * 128;
  const int bb   = blockIdx.y;
  const int wid  = threadIdx.x >> 5;
  const int lane = threadIdx.x & 31;
  const int nsub = wid * 16;

  // stage weights to LDS (2048 x 32B vector copies)
  {
    const v16h* whv = (const v16h*)wh;
    for (int k = threadIdx.x; k < ICH * CIN / 16; k += 256) lw_v[k] = whv[k];
  }

  v8f zero = {};
  v8f acc[8];
  #pragma unroll
  for (int t = 0; t < 8; ++t) acc[t] = zero;

  for (int c0 = 0; c0 < CIN; c0 += 32) {
    // stage activation chunk: float4 (b128) global loads, scatter to (n,c) LDS
    #pragma unroll
    for (int k = threadIdx.x; k < 32 * 128 / 4; k += 256) {
      int c  = k >> 5;
      int n4 = (k & 31) * 4;
      size_t gi = ((size_t)bb * CIN + c0 + c) * HW + n0 + n4;
      float4 v = *(const float4*)(x + gi);
      if (c0 + 32 < CIN) __builtin_prefetch(x + gi + (size_t)32 * HW, 0, 0);
      lx[(n4 + 0) * LDP + c] = (_Float16)v.x;
      lx[(n4 + 1) * LDP + c] = (_Float16)v.y;
      lx[(n4 + 2) * LDP + c] = (_Float16)v.z;
      lx[(n4 + 3) * LDP + c] = (_Float16)v.w;
    }
    __syncthreads();
    v16h a = afrag_rm(lx + nsub * LDP, LDP, lane);   // 2x ds_load_b128
    v16h bf[8];
    #pragma unroll
    for (int t = 0; t < 8; ++t)
      bf[t] = bfrag_nm(lw + (size_t)(t * 16) * CIN + c0, CIN, lane);
    #pragma unroll
    for (int t = 0; t < 8; ++t)
      acc[t] = wmma16(a, bf[t], acc[t]);
    __syncthreads();
  }
  const int col  = lane & 15;
  const int rofs = (lane >> 4) * 8;
  for (int t = 0; t < 8; ++t) {
    int i = t * 16 + col;
    float bi = bias[i];
    #pragma unroll
    for (int r = 0; r < 8; ++r) {
      size_t o = ((size_t)bb * HW + n0 + nsub + rofs + r) * ICH + i;
      outm[o] = (_Float16)(acc[t][r] + bi);
    }
  }
}

// ---------------- 2x2 max-pool: g -> (ic, m), phi -> (m, ic) ----------------
__global__ __launch_bounds__(256) void pool_kernel(
    const _Float16* __restrict__ gfull, const _Float16* __restrict__ pfull,
    _Float16* __restrict__ gp, _Float16* __restrict__ pp) {
  int t = blockIdx.x * 256 + threadIdx.x;
  if (t >= NB * MP * ICH) return;
  int i  = t & (ICH - 1);
  int m  = (t >> 7) & (MP - 1);
  int bb = t >> 17;
  int py = m >> 5, px = m & 31;
  int n  = (py * 2) * 64 + px * 2;
  size_t i0 = ((size_t)bb * HW + n) * ICH + i;
  float g0 = (float)gfull[i0],            g1 = (float)gfull[i0 + ICH];
  float g2 = (float)gfull[i0 + 64 * ICH], g3 = (float)gfull[i0 + 65 * ICH];
  float p0 = (float)pfull[i0],            p1 = (float)pfull[i0 + ICH];
  float p2 = (float)pfull[i0 + 64 * ICH], p3 = (float)pfull[i0 + 65 * ICH];
  gp[((size_t)bb * ICH + i) * MP + m] = (_Float16)fmaxf(fmaxf(g0, g1), fmaxf(g2, g3));
  pp[((size_t)bb * MP + m) * ICH + i] = (_Float16)fmaxf(fmaxf(p0, p1), fmaxf(p2, p3));
}

// ---------------- fused attention: f = theta*phi^T, softmax, y = P*g^T ------
__global__ __launch_bounds__(256) void attn_kernel(
    const _Float16* __restrict__ theta, const _Float16* __restrict__ pp,
    const _Float16* __restrict__ gp, _Float16* __restrict__ y) {
  __shared__ float    f[16 * MP];    // 64 KB fp32 score tile
  __shared__ _Float16 pf[16 * MP];   // 32 KB normalized probs (f16)
  const int row0 = blockIdx.x * 16;
  const int bb   = blockIdx.y;
  const int wid  = threadIdx.x >> 5;
  const int lane = threadIdx.x & 31;
  const int m0   = wid * 128;

  v8f zero = {};
  v8f acc[8];
  #pragma unroll
  for (int t = 0; t < 8; ++t) acc[t] = zero;

  const _Float16* arow = theta + ((size_t)bb * HW + row0) * ICH;
  #pragma unroll
  for (int k0 = 0; k0 < ICH; k0 += 32) {
    v16h a = afrag_rm(arow + k0, ICH, lane);
    v16h bf[8];
    #pragma unroll
    for (int t = 0; t < 8; ++t)
      bf[t] = bfrag_nm(pp + ((size_t)bb * MP + m0 + t * 16) * ICH + k0, ICH, lane);
    #pragma unroll
    for (int t = 0; t < 8; ++t)
      acc[t] = wmma16(a, bf[t], acc[t]);
  }
  const int col  = lane & 15;
  const int rofs = (lane >> 4) * 8;
  for (int t = 0; t < 8; ++t) {
    #pragma unroll
    for (int r = 0; r < 8; ++r)
      f[(rofs + r) * MP + m0 + t * 16 + col] = acc[t][r];
  }
  __syncthreads();

  // softmax: each wave owns 2 rows; wave32 shuffle reductions; emit f16 probs
  for (int rr = 0; rr < 2; ++rr) {
    int row = wid * 2 + rr;
    float*    fr = f + row * MP;
    _Float16* pr = pf + row * MP;
    float mx = -1e30f;
    for (int c = lane; c < MP; c += 32) mx = fmaxf(mx, fr[c]);
    for (int o = 16; o >= 1; o >>= 1) mx = fmaxf(mx, __shfl_xor(mx, o, 32));
    float s = 0.f;
    for (int c = lane; c < MP; c += 32) {
      float e = __expf(fr[c] - mx);
      fr[c] = e;
      s += e;
    }
    for (int o = 16; o >= 1; o >>= 1) s += __shfl_xor(s, o, 32);
    float inv = 1.f / s;
    for (int c = lane; c < MP; c += 32) pr[c] = (_Float16)(fr[c] * inv);
  }
  __syncthreads();

  // y(row, i) = sum_m P[row][m] * g[i][m] ; each wave: one 16-wide i tile.
  // P is f16 row-major in LDS -> A fragment is two ds_load_b128.
  v8f acy = zero;
  const _Float16* gbase = gp + ((size_t)bb * ICH + wid * 16) * MP;
  for (int k0 = 0; k0 < MP; k0 += 32) {
    v16h a = afrag_rm(pf + k0, MP, lane);
    v16h b = bfrag_nm(gbase + k0, MP, lane);
    acy = wmma16(a, b, acy);
  }
  #pragma unroll
  for (int r = 0; r < 8; ++r)
    y[((size_t)bb * HW + row0 + rofs + r) * ICH + wid * 16 + col] = (_Float16)acy[r];
}

// ---------------- final conv: z = W*y^T + bias + residual ----------------
__global__ __launch_bounds__(256) void out_kernel(
    const _Float16* __restrict__ whW, const float* __restrict__ Wb,
    const _Float16* __restrict__ y, const float* __restrict__ xtt,
    float* __restrict__ out) {
  const int n0   = blockIdx.x * 64;
  const int bb   = blockIdx.y;
  const int wid  = threadIdx.x >> 5;
  const int lane = threadIdx.x & 31;
  v8f zero = {};
  v8f acc[2][4];
  #pragma unroll
  for (int h = 0; h < 2; ++h)
    #pragma unroll
    for (int nt = 0; nt < 4; ++nt) acc[h][nt] = zero;

  #pragma unroll
  for (int k0 = 0; k0 < ICH; k0 += 32) {
    v16h a0 = afrag_rm(whW + (size_t)(wid * 16) * ICH + k0, ICH, lane);
    v16h a1 = afrag_rm(whW + (size_t)((wid + 8) * 16) * ICH + k0, ICH, lane);
    v16h bf[4];
    #pragma unroll
    for (int nt = 0; nt < 4; ++nt)
      bf[nt] = bfrag_nm(y + ((size_t)bb * HW + n0 + nt * 16) * ICH + k0, ICH, lane);
    #pragma unroll
    for (int nt = 0; nt < 4; ++nt) {
      acc[0][nt] = wmma16(a0, bf[nt], acc[0][nt]);
      acc[1][nt] = wmma16(a1, bf[nt], acc[1][nt]);
    }
  }
  const int col  = lane & 15;
  const int rofs = (lane >> 4) * 8;
  for (int h = 0; h < 2; ++h) {
    int obase = (wid + 8 * h) * 16;
    #pragma unroll
    for (int r = 0; r < 8; ++r) {
      int o = obase + rofs + r;
      float bias = Wb[o];
      #pragma unroll
      for (int nt = 0; nt < 4; ++nt) {
        size_t oi = ((size_t)bb * CIN + o) * HW + n0 + nt * 16 + col;
        out[oi] = acc[h][nt][r] + bias + xtt[oi];
      }
    }
  }
}

extern "C" void kernel_launch(void* const* d_in, const int* in_sizes, int n_in,
                              void* d_out, int out_size, void* d_ws, size_t ws_size,
                              hipStream_t stream) {
  (void)in_sizes; (void)n_in; (void)out_size; (void)ws_size;
  const float* base = (const float*)d_in[0];
  const float* tt   = (const float*)d_in[1];
  const float* g_w  = (const float*)d_in[2];
  const float* g_b  = (const float*)d_in[3];
  const float* th_w = (const float*)d_in[4];
  const float* th_b = (const float*)d_in[5];
  const float* ph_w = (const float*)d_in[6];
  const float* ph_b = (const float*)d_in[7];
  const float* W_w  = (const float*)d_in[8];
  const float* W_b  = (const float*)d_in[9];
  float* out = (float*)d_out;

  char* ws = (char*)d_ws;
  const size_t WSZ = (size_t)ICH * CIN * sizeof(_Float16);     // 64 KB
  const size_t ACT = (size_t)NB * HW * ICH * sizeof(_Float16); // 8 MB
  const size_t PSZ = (size_t)NB * MP * ICH * sizeof(_Float16); // 2 MB
  size_t off = 0;
  _Float16* wh_g  = (_Float16*)(ws + off); off += WSZ;
  _Float16* wh_t  = (_Float16*)(ws + off); off += WSZ;
  _Float16* wh_p  = (_Float16*)(ws + off); off += WSZ;
  _Float16* wh_W  = (_Float16*)(ws + off); off += WSZ;
  _Float16* theta = (_Float16*)(ws + off); off += ACT;
  _Float16* gfull = (_Float16*)(ws + off); off += ACT;
  _Float16* pfull = (_Float16*)(ws + off); off += ACT;
  _Float16* gp    = (_Float16*)(ws + off); off += PSZ;
  _Float16* pp    = (_Float16*)(ws + off); off += PSZ;
  _Float16* yb    = (_Float16*)(ws + off); off += ACT;

  cvt_w_kernel<<<dim3(128), dim3(256), 0, stream>>>(g_w, th_w, ph_w, W_w,
                                                    wh_g, wh_t, wh_p, wh_W);
  proj_kernel<<<dim3(32, 8), dim3(256), 0, stream>>>(base, wh_t, th_b, theta);
  proj_kernel<<<dim3(32, 8), dim3(256), 0, stream>>>(tt, wh_g, g_b, gfull);
  proj_kernel<<<dim3(32, 8), dim3(256), 0, stream>>>(tt, wh_p, ph_b, pfull);
  pool_kernel<<<dim3(4096), dim3(256), 0, stream>>>(gfull, pfull, gp, pp);
  attn_kernel<<<dim3(256, 8), dim3(256), 0, stream>>>(theta, pp, gp, yb);
  out_kernel<<<dim3(64, 8), dim3(256), 0, stream>>>(wh_W, W_b, yb, tt, out);
}
